// ResGatedGCN_17721035063717
// MI455X (gfx1250) — compile-verified, matrix operand order checked by reading, not run on hardware
//
#include <hip/hip_runtime.h>
#include <cstddef>
#include <cstdint>

// ---------------------------------------------------------------------------
// ResGatedGraphConv x4 + BatchNorm/ReLU for MI455X (gfx1250, wave32, WMMA)
//
// Per layer:
//   [K|Q|V|S] = x @ [Wk|Wq|Wv|Ws] + [bk|bq|bv|bs]   (bf16 WMMA, f32 accum)
//   agg = S;  agg[dst] += sigmoid(K[dst]+Q[src]) * V[src]   (f32 atomics)
//   x' = relu(batchnorm(agg)) -> bf16                (layers 0..2)
// ---------------------------------------------------------------------------

typedef __attribute__((ext_vector_type(16))) __bf16 v16bf;
typedef __attribute__((ext_vector_type(8)))  __bf16 v8bf;
typedef __attribute__((ext_vector_type(4)))  __bf16 v4bf;
typedef __attribute__((ext_vector_type(8)))  float  v8f;

#define HID     128
#define OUTD    64
#define BN_EPS  1e-5f

// -------------------------------- f32 -> bf16 cast of node features --------
__global__ void cast_f32_bf16(const float* __restrict__ x, __bf16* __restrict__ xb,
                              int total4) {
  int i = blockIdx.x * blockDim.x + threadIdx.x;
  if (i >= total4) return;
  const float4 v = *(const float4*)(x + (size_t)i * 4);
  v4bf o;
  o[0] = (__bf16)v.x; o[1] = (__bf16)v.y; o[2] = (__bf16)v.z; o[3] = (__bf16)v.w;
  *(v4bf*)(xb + (size_t)i * 4) = o;
}

// -------- pack weights into WMMA-B-fragment-major bf16 layout + fused bias --
// Packed element index: ((ks*nct + ct)*32 + lane)*16 + i  holds
//   W[K][c] with K = ks*32 + i + 8*((i>>3) + (lane>>4)),  c = ct*16 + (lane&15)
__global__ void pack_weights(const float* __restrict__ Wk, const float* __restrict__ bk,
                             const float* __restrict__ Wq, const float* __restrict__ bq,
                             const float* __restrict__ Wv, const float* __restrict__ bv,
                             const float* __restrict__ Ws, const float* __restrict__ bs,
                             __bf16* __restrict__ pw, float* __restrict__ bcat, int dout) {
  const int D4  = 4 * dout;
  const int nct = D4 / 16;
  const int total = 128 * D4;            // 4 ksteps * nct * 32 lanes * 16 elems
  for (int idx = blockIdx.x * blockDim.x + threadIdx.x; idx < total + D4;
       idx += gridDim.x * blockDim.x) {
    if (idx < total) {
      const int i    = idx & 15;
      const int lane = (idx >> 4) & 31;
      const int ct   = (idx >> 9) % nct;
      const int ks   = idx / (nct << 9);
      const int K    = ks * 32 + i + 8 * ((i >> 3) + (lane >> 4));
      const int c    = ct * 16 + (lane & 15);
      const int blk  = c / dout, wc = c % dout;
      const float* W = (blk == 0) ? Wk : (blk == 1) ? Wq : (blk == 2) ? Wv : Ws;
      pw[idx] = (__bf16)W[(size_t)K * dout + wc];
    } else {
      const int c   = idx - total;
      const int blk = c / dout, wc = c % dout;
      const float* b = (blk == 0) ? bk : (blk == 1) ? bq : (blk == 2) ? bv : bs;
      bcat[c] = b[wc];
    }
  }
}

// ------------------------- fused K|Q|V|S GEMM via v_wmma_f32_16x16x32_bf16 --
// One block = 16 output rows, 8 waves; wave w owns col tiles {w, w+8, ...}.
template <int CPW>   // col-tiles per wave: 4 (dout=128) or 2 (dout=64)
__global__ void gemm_kqvs(const __bf16* __restrict__ xb, const __bf16* __restrict__ pw,
                          const float* __restrict__ bcat,
                          float* __restrict__ kqv, float* __restrict__ agg,
                          int N, int dout) {
  const int nct  = (4 * dout) / 16;
  const int rt   = blockIdx.x;
  const int tid  = threadIdx.x;
  const int wave = tid >> 5;
  const int lane = tid & 31;
  const int half = lane >> 4;
  const int m    = lane & 15;

  __shared__ __align__(16) __bf16 lds_a[16 * 128];   // 16 rows x 128 (bf16)

  // Stage A tile: 256 threads x 16 bytes (rows clamped for safety).
  {
    int row  = tid >> 4;                 // 0..15
    int seg  = tid & 15;                 // 16-byte segment within a 256B row
    int grow = rt * 16 + row;
    if (grow >= N) grow = 0;             // N % 16 == 0 in this workload
    const uint4* src = (const uint4*)(xb + (size_t)grow * 128) + seg;
    ((uint4*)lds_a)[row * 16 + seg] = *src;
  }
  __syncthreads();

  const v8f zero = {0.f, 0.f, 0.f, 0.f, 0.f, 0.f, 0.f, 0.f};
  v8f acc[CPW];
#pragma unroll
  for (int j = 0; j < CPW; ++j) acc[j] = zero;

#pragma unroll
  for (int ks = 0; ks < 4; ++ks) {       // K = 128 in steps of 32
    // A fragment: lane holds row m, K runs [ks*32 + 8*half .. +8) and (+16..)
    const __bf16* ab = lds_a + m * 128 + ks * 32 + half * 8;
    const v8bf alo = *(const v8bf*)ab;
    const v8bf ahi = *(const v8bf*)(ab + 16);
    const v16bf a = __builtin_shufflevector(alo, ahi,
        0, 1, 2, 3, 4, 5, 6, 7, 8, 9, 10, 11, 12, 13, 14, 15);
#pragma unroll
    for (int j = 0; j < CPW; ++j) {
      const int ct = wave + j * 8;
      const __bf16* bb = pw + ((size_t)((ks * nct + ct) * 32 + lane)) * 16;
      const v8bf blo = *(const v8bf*)bb;
      const v8bf bhi = *(const v8bf*)(bb + 8);
      const v16bf b = __builtin_shufflevector(blo, bhi,
          0, 1, 2, 3, 4, 5, 6, 7, 8, 9, 10, 11, 12, 13, 14, 15);
      acc[j] = __builtin_amdgcn_wmma_f32_16x16x32_bf16(
          false, a, false, b, (short)0, acc[j], false, false);
    }
  }

  // Epilogue: D layout -> row = rt*16 + r + 8*half, col = ct*16 + (lane&15).
  const int n       = lane & 15;
  const int rowbase = rt * 16 + 8 * half;
  const int kqvw    = 3 * dout;
#pragma unroll
  for (int j = 0; j < CPW; ++j) {
    const int ct = wave + j * 8;
    const int c  = ct * 16 + n;
    const float bias = bcat[c];
    const bool  isS  = (c >= kqvw);
#pragma unroll
    for (int r = 0; r < 8; ++r) {
      const int row = rowbase + r;
      if (row >= N) continue;
      const float v = acc[j][r] + bias;
      if (isS) agg[(size_t)row * dout + (c - kqvw)] = v;
      else     kqv[(size_t)row * kqvw + c] = v;
    }
  }
}

// ---------------- edge kernel: gather, sigmoid gate, scatter-add (f32) ------
__global__ void edge_messages(const int* __restrict__ ei, const float* __restrict__ kqv,
                              float* __restrict__ agg, int E, int dout) {
  const int e    = blockIdx.x * (blockDim.x >> 5) + (threadIdx.x >> 5);
  const int lane = threadIdx.x & 31;
  if (e >= E) return;
  const int src = ei[e];
  const int dst = ei[E + e];
  const int c   = lane * 4;
  if (c >= dout) return;
  const int kqvw = 3 * dout;
  const float4 k = *(const float4*)(kqv + (size_t)dst * kqvw + c);
  const float4 q = *(const float4*)(kqv + (size_t)src * kqvw + dout + c);
  const float4 v = *(const float4*)(kqv + (size_t)src * kqvw + 2 * dout + c);
  float4 g;
  g.x = v.x / (1.f + __expf(-(k.x + q.x)));
  g.y = v.y / (1.f + __expf(-(k.y + q.y)));
  g.z = v.z / (1.f + __expf(-(k.z + q.z)));
  g.w = v.w / (1.f + __expf(-(k.w + q.w)));
  float* out = agg + (size_t)dst * dout + c;
  unsafeAtomicAdd(out + 0, g.x);
  unsafeAtomicAdd(out + 1, g.y);
  unsafeAtomicAdd(out + 2, g.z);
  unsafeAtomicAdd(out + 3, g.w);
}

// ------------------------- BatchNorm (training stats) + ReLU + bf16 cast ----
__global__ void bn_stats(const float* __restrict__ agg, float* __restrict__ stats, int N) {
  const int c = threadIdx.x;  // 128 features
  float s = 0.f, sq = 0.f;
  for (int r = blockIdx.x; r < N; r += gridDim.x) {
    const float v = agg[(size_t)r * HID + c];
    s += v; sq += v * v;
  }
  unsafeAtomicAdd(&stats[c], s);
  unsafeAtomicAdd(&stats[HID + c], sq);
}

__global__ void bn_finalize(float* __restrict__ stats, const float* __restrict__ gamma,
                            const float* __restrict__ beta, int N) {
  const int c  = threadIdx.x;
  const float invN = 1.f / (float)N;
  const float mu   = stats[c] * invN;
  const float var  = stats[HID + c] * invN - mu * mu;
  const float sc   = gamma[c] * rsqrtf(var + BN_EPS);
  stats[2 * HID + c] = sc;
  stats[3 * HID + c] = beta[c] - mu * sc;
}

__global__ void bn_apply_relu_cast(const float* __restrict__ agg,
                                   const float* __restrict__ stats,
                                   __bf16* __restrict__ xb, int total4) {
  const int i = blockIdx.x * blockDim.x + threadIdx.x;
  if (i >= total4) return;
  const int c4 = (i & 31) * 4;                     // 128 features / 4
  const float4 v = *(const float4*)(agg + (size_t)i * 4);
  v4bf o;
  float y;
  y = v.x * stats[2 * HID + c4 + 0] + stats[3 * HID + c4 + 0]; o[0] = (__bf16)fmaxf(y, 0.f);
  y = v.y * stats[2 * HID + c4 + 1] + stats[3 * HID + c4 + 1]; o[1] = (__bf16)fmaxf(y, 0.f);
  y = v.z * stats[2 * HID + c4 + 2] + stats[3 * HID + c4 + 2]; o[2] = (__bf16)fmaxf(y, 0.f);
  y = v.w * stats[2 * HID + c4 + 3] + stats[3 * HID + c4 + 3]; o[3] = (__bf16)fmaxf(y, 0.f);
  *(v4bf*)(xb + (size_t)i * 4) = o;
}

// ---------------------------------------------------------------------------
static inline size_t align256(size_t v) { return (v + 255) & ~(size_t)255; }

extern "C" void kernel_launch(void* const* d_in, const int* in_sizes, int n_in,
                              void* d_out, int out_size, void* d_ws, size_t ws_size,
                              hipStream_t stream) {
  const float* x  = (const float*)d_in[0];
  const int*   ei = (const int*)d_in[1];
  const int N = in_sizes[0] / HID;       // 100000
  const int E = in_sizes[1] / 2;         // 600000

  // Input layout (setup_inputs insertion order):
  //   2 + l*8 + {0:key.W,1:key.b,2:query.W,3:query.b,4:value.W,5:value.b,6:skip.W,7:skip.b}
  //   34 + i*2: bns[i].gamma ; 35 + i*2: bns[i].beta
  // Workspace carve-up.
  char* ws = (char*)d_ws;
  __bf16* xb   = (__bf16*)ws;  ws += align256((size_t)N * HID * sizeof(__bf16));
  float*  kqv  = (float*)ws;   ws += align256((size_t)N * 3 * HID * sizeof(float));
  float*  agg  = (float*)ws;   ws += align256((size_t)N * HID * sizeof(float));
  __bf16* pw   = (__bf16*)ws;  ws += align256((size_t)HID * 4 * HID * sizeof(__bf16));
  float*  bcat = (float*)ws;   ws += align256(4 * HID * sizeof(float));
  float*  stats = (float*)ws;  ws += align256(4 * HID * sizeof(float));

  // x (f32) -> xb (bf16)
  {
    const int total4 = (N * HID) / 4;
    cast_f32_bf16<<<(total4 + 255) / 256, 256, 0, stream>>>(x, xb, total4);
  }

  const int rowTiles = (N + 15) / 16;
  for (int l = 0; l < 4; ++l) {
    const int dout = (l == 3) ? OUTD : HID;
    const int D4   = 4 * dout;
    const float* Wk = (const float*)d_in[2 + l * 8 + 0];
    const float* bk = (const float*)d_in[2 + l * 8 + 1];
    const float* Wq = (const float*)d_in[2 + l * 8 + 2];
    const float* bq = (const float*)d_in[2 + l * 8 + 3];
    const float* Wv = (const float*)d_in[2 + l * 8 + 4];
    const float* bv = (const float*)d_in[2 + l * 8 + 5];
    const float* Wsk = (const float*)d_in[2 + l * 8 + 6];
    const float* bsk = (const float*)d_in[2 + l * 8 + 7];

    // Pack weights into fragment-major bf16 + concatenated bias.
    {
      const int total = 128 * D4 + D4;
      pack_weights<<<(total + 255) / 256, 256, 0, stream>>>(
          Wk, bk, Wq, bq, Wv, bv, Wsk, bsk, pw, bcat, dout);
    }

    float* aggp = (l == 3) ? (float*)d_out : agg;

    // Fused K|Q|V|S GEMM (WMMA bf16); skip-term lands directly in agg/d_out.
    if (dout == HID)
      gemm_kqvs<4><<<rowTiles, 256, 0, stream>>>(xb, pw, bcat, kqv, aggp, N, dout);
    else
      gemm_kqvs<2><<<rowTiles, 256, 0, stream>>>(xb, pw, bcat, kqv, aggp, N, dout);

    // Gather / gate / scatter-add (1 wave per edge, 8 edges per block).
    edge_messages<<<(E + 7) / 8, 256, 0, stream>>>(ei, kqv, aggp, E, dout);

    if (l < 3) {
      const float* gamma = (const float*)d_in[34 + l * 2];
      const float* beta  = (const float*)d_in[35 + l * 2];
      hipMemsetAsync(stats, 0, 4 * HID * sizeof(float), stream);
      bn_stats<<<512, HID, 0, stream>>>(agg, stats, N);
      bn_finalize<<<1, HID, 0, stream>>>(stats, gamma, beta, N);
      const int total4 = (N * HID) / 4;
      bn_apply_relu_cast<<<(total4 + 255) / 256, 256, 0, stream>>>(agg, stats, xb, total4);
    }
  }
}
